// Seq2SeqLSTMWithAttention_72310069395728
// MI455X (gfx1250) — matrix operand order repaired and checked by reading, not compiled, not run on hardware
//
#include <hip/hip_runtime.h>

// ---------------------------------------------------------------------------
// Types for CDNA5 WMMA (wave32, 16x16x32 bf16 -> f32) and TDM descriptors
// ---------------------------------------------------------------------------
typedef __attribute__((ext_vector_type(16))) __bf16 bf16x16;
typedef __attribute__((ext_vector_type(8)))  __bf16 bf16x8;
typedef __attribute__((ext_vector_type(8)))  float  f32x8;
typedef __attribute__((ext_vector_type(4)))  unsigned int u32x4;
typedef __attribute__((ext_vector_type(8)))  int i32x8;
typedef __attribute__((ext_vector_type(4)))  int i32x4;

#define BB    64      // batch
#define LSEQ  64      // Ls == Lt
#define EMB   512
#define HID   512
#define G4    2048    // 4*H
#define VOUT  32000
#define NROW  4096    // LSEQ * BB

#define NBLK  32      // recurrence kernels: blocks
#define NTHR  256     // threads per block (8 waves)

// ---------------------------------------------------------------------------
// Fragment helpers (layouts per CDNA5 ISA 7.12.2)
// ---------------------------------------------------------------------------
__device__ __forceinline__ f32x8 f8zero() {
  f32x8 v;
#pragma unroll
  for (int i = 0; i < 8; ++i) v[i] = 0.f;
  return v;
}

// A: 16x32 bf16 tile from row-major matrix (lda multiple of 8, 16B aligned).
// lane<16:  row=lane,    K = {0..7, 16..23}
// lane>=16: row=lane-16, K = {8..15, 24..31}
__device__ __forceinline__ bf16x16 load_a_frag(const __bf16* __restrict__ A,
                                               int lda, int lane) {
  int r  = lane & 15;
  int kb = (lane & 16) ? 8 : 0;
  const __bf16* p = A + (size_t)r * lda + kb;
  bf16x8 lo = *(const bf16x8*)(p);
  bf16x8 hi = *(const bf16x8*)(p + 16);
  bf16x16 a;
#pragma unroll
  for (int i = 0; i < 8; ++i) { a[i] = lo[i]; a[i + 8] = hi[i]; }
  return a;
}

// B: pre-swizzled tile, 32 lanes x 16 bf16 contiguous per lane.
// Works for both global and LDS-resident tiles.
__device__ __forceinline__ bf16x16 load_b_frag(const __bf16* P, int lane) {
  const __bf16* p = P + (size_t)lane * 16;
  bf16x8 lo = *(const bf16x8*)(p);
  bf16x8 hi = *(const bf16x8*)(p + 8);
  bf16x16 b;
#pragma unroll
  for (int i = 0; i < 8; ++i) { b[i] = lo[i]; b[i + 8] = hi[i]; }
  return b;
}

// C/D: VGPR i: lanes 0-15 -> M=i, N=lane; lanes 16-31 -> M=i+8, N=lane-16.
__device__ __forceinline__ f32x8 load_c_frag(const float* __restrict__ C,
                                             int ldc, int lane) {
  int n  = lane & 15;
  int mo = (lane & 16) ? 8 : 0;
  f32x8 v;
#pragma unroll
  for (int i = 0; i < 8; ++i) v[i] = C[(size_t)(mo + i) * ldc + n];
  return v;
}

__device__ __forceinline__ void store_c_frag(float* __restrict__ C, int ldc,
                                             int lane, f32x8 v) {
  int n  = lane & 15;
  int mo = (lane & 16) ? 8 : 0;
#pragma unroll
  for (int i = 0; i < 8; ++i) C[(size_t)(mo + i) * ldc + n] = v[i];
}

__device__ __forceinline__ f32x8 wmma_bf16(bf16x16 a, bf16x16 b, f32x8 c) {
  return __builtin_amdgcn_wmma_f32_16x16x32_bf16(false, a, false, b,
                                                 (short)0, c, false, false);
}

__device__ __forceinline__ float sigmoidf(float x) {
  return 1.f / (1.f + expf(-x));
}

__device__ __forceinline__ float wave_max(float v) {
#pragma unroll
  for (int o = 16; o > 0; o >>= 1) v = fmaxf(v, __shfl_xor(v, o, 32));
  return v;
}
__device__ __forceinline__ float wave_sum(float v) {
#pragma unroll
  for (int o = 16; o > 0; o >>= 1) v += __shfl_xor(v, o, 32);
  return v;
}

// Software grid barrier (32 resident blocks). Generation read BEFORE arrival.
__device__ __forceinline__ void grid_sync(int* cnt, int* gen) {
  __threadfence();
  __syncthreads();
  if (threadIdx.x == 0) {
    int g = atomicAdd(gen, 0);
    int v = atomicAdd(cnt, 1);
    if (v == NBLK - 1) {
      atomicExch(cnt, 0);
      __threadfence();
      atomicAdd(gen, 1);
    } else {
      while (atomicAdd(gen, 0) == g) __builtin_amdgcn_s_sleep(2);
    }
  }
  __syncthreads();
  __threadfence();
}

// ---------------------------------------------------------------------------
// Tensor Data Mover: stage one B panel chunk (4 n-tiles x 512 bf16 = 4KB)
// into LDS. 2D tile of 8-byte elements: tile_dim0=128 units (1KB row),
// tile_dim1=4 rows, row stride = KT*128 units (= KT*512 bf16).
// Descriptor layout per CDNA5 ISA ch.8 (D# groups 0/1; groups 2/3 zero).
// This toolchain exposes the 6-arg builtin:
//   (u32x4 g0, i32x8 g1, i32x4, i32x4, i32x8, i32 cpol)
// ---------------------------------------------------------------------------
__device__ __forceinline__ void tdm_load_b_panel(const __bf16* gptr,
                                                 unsigned lds_off, int KT) {
  unsigned long long ga = (unsigned long long)(uintptr_t)gptr;
  u32x4 g0;
  g0[0] = 1u;                                  // count=1, user descriptor
  g0[1] = lds_off;                             // LDS byte address
  g0[2] = (unsigned)(ga & 0xFFFFFFFFu);        // global_addr[31:0]
  g0[3] = (unsigned)((ga >> 32) & 0x01FFFFFFu) // global_addr[56:32]
          | (2u << 30);                        // type = 2 ("image")
  unsigned d0s = (unsigned)KT * 128u;          // dim0 stride, 8B units
  i32x8 g1;
  g1[0] = (int)(3u << 16);                     // data_size = 3 (8 bytes)
  g1[1] = (int)((d0s & 0xFFFFu) << 16);        // tensor_dim0[15:0]
  g1[2] = (int)((d0s >> 16) | (2048u << 16));  // tensor_dim0[31:16]|dim1[15:0]
  g1[3] = (int)(128u << 16);                   // dim1[31:16]=0 | tile_dim0=128
  g1[4] = 4;                                   // tile_dim1 = 4 (tile_dim2 = 0)
  g1[5] = (int)d0s;                            // tensor_dim0_stride[31:0]
  g1[6] = 0;                                   // stride0 hi | stride1 lo
  g1[7] = 0;                                   // stride1 hi
  i32x4 z4;
  z4[0] = 0; z4[1] = 0; z4[2] = 0; z4[3] = 0;
  i32x8 z8;
#pragma unroll
  for (int i = 0; i < 8; ++i) z8[i] = 0;
  __builtin_amdgcn_tensor_load_to_lds(g0, g1, z4, z4, z8, 0);
}

// ---------------------------------------------------------------------------
// Weight packing: row-major f32 (K x N, leading dim ldW, row offset) ->
// bf16 B-fragment tiles [ntile][ktile][lane][16].
// ---------------------------------------------------------------------------
__global__ void pack_b_kernel(const float* __restrict__ W, __bf16* __restrict__ P,
                              int KT, int ldW, int row_off, int total) {
  int idx = blockIdx.x * 256 + threadIdx.x;
  if (idx >= total) return;
  int tile   = idx >> 9;
  int within = idx & 511;
  int lane   = within >> 4;
  int i      = within & 15;
  int ntile  = tile / KT;
  int ktile  = tile - ntile * KT;
  int n = ntile * 16 + (lane & 15);
  int k = ktile * 32 + ((lane & 16) ? 8 : 0) + (i & 7) + ((i & 8) ? 16 : 0);
  P[idx] = (__bf16)W[(size_t)(row_off + k) * ldW + n];
}

// Embedding gather -> bf16 rows ordered r = t*B + b.
__global__ void gather_embed_kernel(const int* __restrict__ tok,
                                    const float* __restrict__ table,
                                    __bf16* __restrict__ out) {
  int r  = blockIdx.x;
  int b  = r & 63;
  int t  = r >> 6;
  int tk = tok[b * LSEQ + t];
  const float* src = table + (size_t)tk * EMB;
  __bf16* dst = out + (size_t)r * EMB;
  for (int e = threadIdx.x; e < EMB; e += 128) dst[e] = (__bf16)src[e];
}

// ---------------------------------------------------------------------------
// Generic bf16 WMMA GEMM with TDM-staged, double-buffered B panels in LDS:
// C(f32, MxN) = A(bf16 MxK row-major) @ Bpacked + bias.
// Block = 8 waves, tile 128x64; per kt the 4KB B chunk is DMA'd once per
// block (wave 0 issues TENSOR_LOAD_TO_LDS, waits TENSORcnt, barrier) and
// consumed by all 8 waves from LDS -> 8x less L2/global B traffic.
// REMAP: route rows (t*B+b) -> out row (b*Lt+t) into d_out.
// ---------------------------------------------------------------------------
template <bool REMAP>
__global__ void __launch_bounds__(256)
gemm_bf16_kernel(const __bf16* __restrict__ A, const __bf16* __restrict__ Bp,
                 const float* __restrict__ bias, float* __restrict__ C,
                 int K, int N) {
  __shared__ __attribute__((aligned(16))) __bf16 bstage[2][2048]; // 2 x 4KB
  const int KT   = K >> 5;
  const int wave = threadIdx.x >> 5;
  const int lane = threadIdx.x & 31;
  const int m0   = blockIdx.x * 128 + wave * 16;
  const int nt0  = blockIdx.y * 4;  // first of 4 n-tiles
  const __bf16* Arow   = A + (size_t)m0 * K;
  const __bf16* Bpanel = Bp + (size_t)nt0 * KT * 512; // row j: ntile nt0+j

  // Preload kt=0 panel.
  if (wave == 0) {
    tdm_load_b_panel(Bpanel, (unsigned)(uintptr_t)&bstage[0][0], KT);
    __builtin_amdgcn_s_wait_tensorcnt(0);
  }
  __syncthreads();

  f32x8 acc[4];
#pragma unroll
  for (int j = 0; j < 4; ++j) acc[j] = f8zero();

  for (int kt = 0; kt < KT; ++kt) {
    // Kick DMA for next panel into the other buffer.
    if (wave == 0 && kt + 1 < KT)
      tdm_load_b_panel(Bpanel + (size_t)(kt + 1) * 512,
                       (unsigned)(uintptr_t)&bstage[(kt + 1) & 1][0], KT);

    bf16x16 a = load_a_frag(Arow + kt * 32, K, lane);
    const __bf16* bb = &bstage[kt & 1][0];
#pragma unroll
    for (int j = 0; j < 4; ++j) {
      bf16x16 b = load_b_frag(bb + j * 512, lane);
      acc[j] = wmma_bf16(a, b, acc[j]);
    }

    if (wave == 0 && kt + 1 < KT) __builtin_amdgcn_s_wait_tensorcnt(0);
    __syncthreads();
  }

  const int n_in = lane & 15;
  const int mo   = (lane & 16) ? 8 : 0;
#pragma unroll
  for (int j = 0; j < 4; ++j) {
    int n = (nt0 + j) * 16 + n_in;
    float bv = bias ? bias[n] : 0.f;
#pragma unroll
    for (int i = 0; i < 8; ++i) {
      float v = acc[j][i] + bv;
      int m = m0 + mo + i;
      if (REMAP) {
        int b_ = m & 63, t_ = m >> 6;
        C[(size_t)(b_ * LSEQ + t_) * VOUT + n] = v;
      } else {
        C[(size_t)m * N + n] = v;
      }
    }
  }
}

// ---------------------------------------------------------------------------
// Encoder recurrence: 32 blocks x 256 threads, loops t = 0..63.
// z = Xpre[t] + h_bf16 @ WhP ; gates ; store h (f32 + bf16).
// ---------------------------------------------------------------------------
__global__ void __launch_bounds__(256)
encoder_kernel(const float* __restrict__ Xpre, const __bf16* __restrict__ WhP,
               float* __restrict__ c, __bf16* __restrict__ hbf,
               float* __restrict__ z, float* __restrict__ encHidF,
               __bf16* __restrict__ encHidB, int* cnt, int* gen) {
  const int tid  = threadIdx.x;
  const int lane = tid & 31;
  const int gw   = blockIdx.x * 8 + (tid >> 5);   // 0..255
  const int gtid = blockIdx.x * NTHR + tid;       // 0..8191

  // zero h0, c0
  for (int i = gtid; i < BB * HID; i += NBLK * NTHR) {
    c[i] = 0.f; hbf[i] = (__bf16)0.f;
  }
  grid_sync(cnt, gen);

  const int m0  = (gw & 3) * 16;   // 4 m-tiles over B=64
  const int nb  = gw >> 2;         // 0..63 -> two n-tiles: nb, nb+64

#pragma unroll 1
  for (int t = 0; t < LSEQ; ++t) {
    const float* Xt = Xpre + (size_t)t * BB * G4;
    f32x8 acc0 = load_c_frag(Xt + (size_t)m0 * G4 + nb * 16, G4, lane);
    f32x8 acc1 = load_c_frag(Xt + (size_t)m0 * G4 + (nb + 64) * 16, G4, lane);
#pragma unroll 1
    for (int kt = 0; kt < 16; ++kt) {
      bf16x16 a  = load_a_frag(hbf + (size_t)m0 * HID + kt * 32, HID, lane);
      bf16x16 b0 = load_b_frag(WhP + ((size_t)nb * 16 + kt) * 512, lane);
      bf16x16 b1 = load_b_frag(WhP + ((size_t)(nb + 64) * 16 + kt) * 512, lane);
      acc0 = wmma_bf16(a, b0, acc0);
      acc1 = wmma_bf16(a, b1, acc1);
    }
    store_c_frag(z + (size_t)m0 * G4 + nb * 16, G4, lane, acc0);
    store_c_frag(z + (size_t)m0 * G4 + (nb + 64) * 16, G4, lane, acc1);
    grid_sync(cnt, gen);

    for (int idx = gtid; idx < BB * HID; idx += NBLK * NTHR) {
      int b_ = idx >> 9, j = idx & 511;
      const float* zr = z + (size_t)b_ * G4;
      float zi = zr[j], zf = zr[512 + j], zg = zr[1024 + j], zo = zr[1536 + j];
      float cn = sigmoidf(zf) * c[idx] + sigmoidf(zi) * tanhf(zg);
      float hn = sigmoidf(zo) * tanhf(cn);
      c[idx] = cn;
      hbf[idx] = (__bf16)hn;
      size_t r = (size_t)(t * BB + b_) * HID + j;
      encHidF[r] = hn;
      encHidB[r] = (__bf16)hn;
    }
    grid_sync(cnt, gen);
  }
}

// ---------------------------------------------------------------------------
// Decoder recurrence with Bahdanau attention: 32 blocks x 256 threads.
// ---------------------------------------------------------------------------
__global__ void __launch_bounds__(256)
decoder_kernel(const float* __restrict__ Xpre, const __bf16* __restrict__ WxBotP,
               const __bf16* __restrict__ WhP, const __bf16* __restrict__ WsP,
               const float* __restrict__ encT, const float* __restrict__ encHidF,
               const float* __restrict__ attv, float* __restrict__ c,
               __bf16* __restrict__ hbf, float* __restrict__ z,
               float* __restrict__ decT, float* __restrict__ attn,
               __bf16* __restrict__ ctxbf, __bf16* __restrict__ Hcat,
               int* cnt, int* gen) {
  const int tid  = threadIdx.x;
  const int lane = tid & 31;
  const int gw   = blockIdx.x * 8 + (tid >> 5);
  const int gtid = blockIdx.x * NTHR + tid;

  const int m0 = (gw & 3) * 16;
  const int nb = gw >> 2;

#pragma unroll 1
  for (int t = 0; t < LSEQ; ++t) {
    // ---- A: dec_t = h @ att_Ws  (64x512, K=512) : 128 tiles / 256 waves
    if (gw < 128) {
      int nt = gw >> 2;                     // 0..31
      f32x8 acc = f8zero();
#pragma unroll 1
      for (int kt = 0; kt < 16; ++kt) {
        bf16x16 a = load_a_frag(hbf + (size_t)m0 * HID + kt * 32, HID, lane);
        bf16x16 b = load_b_frag(WsP + ((size_t)nt * 16 + kt) * 512, lane);
        acc = wmma_bf16(a, b, acc);
      }
      store_c_frag(decT + (size_t)m0 * HID + nt * 16, HID, lane, acc);
    }
    grid_sync(cnt, gen);

    // ---- BC: energies + softmax, one wave per batch row
    if (gw < BB) {
      int b_ = gw;
      const float* dt  = decT + (size_t)b_ * HID;
      const float* et0 = encT + ((size_t)lane * BB + b_) * HID;
      const float* et1 = encT + ((size_t)(lane + 32) * BB + b_) * HID;
      float e0 = 0.f, e1 = 0.f;
      for (int hh = 0; hh < HID; ++hh) {
        float dv = dt[hh], av = attv[hh];
        e0 += av * tanhf(et0[hh] + dv);
        e1 += av * tanhf(et1[hh] + dv);
      }
      float m = wave_max(fmaxf(e0, e1));
      float x0 = expf(e0 - m), x1 = expf(e1 - m);
      float inv = 1.f / wave_sum(x0 + x1);
      attn[b_ * 64 + lane]      = x0 * inv;
      attn[b_ * 64 + lane + 32] = x1 * inv;
    }
    grid_sync(cnt, gen);

    // ---- D: context = attn @ enc_hiddens (per batch row)
    for (int idx = gtid; idx < BB * HID; idx += NBLK * NTHR) {
      int b_ = idx >> 9, hh = idx & 511;
      const float* ar = attn + b_ * 64;
      float acc = 0.f;
      for (int s = 0; s < 64; ++s)
        acc += ar[s] * encHidF[((size_t)s * BB + b_) * HID + hh];
      ctxbf[idx] = (__bf16)acc;
      Hcat[(size_t)(t * BB + b_) * (2 * HID) + HID + hh] = (__bf16)acc;
    }
    grid_sync(cnt, gen);

    // ---- E: z = Xpre[t] + ctx @ Wx_bot + h @ Wh
    {
      const float* Xt = Xpre + (size_t)t * BB * G4;
      f32x8 acc0 = load_c_frag(Xt + (size_t)m0 * G4 + nb * 16, G4, lane);
      f32x8 acc1 = load_c_frag(Xt + (size_t)m0 * G4 + (nb + 64) * 16, G4, lane);
#pragma unroll 1
      for (int kt = 0; kt < 16; ++kt) {
        bf16x16 a  = load_a_frag(ctxbf + (size_t)m0 * HID + kt * 32, HID, lane);
        bf16x16 b0 = load_b_frag(WxBotP + ((size_t)nb * 16 + kt) * 512, lane);
        bf16x16 b1 = load_b_frag(WxBotP + ((size_t)(nb + 64) * 16 + kt) * 512, lane);
        acc0 = wmma_bf16(a, b0, acc0);
        acc1 = wmma_bf16(a, b1, acc1);
      }
#pragma unroll 1
      for (int kt = 0; kt < 16; ++kt) {
        bf16x16 a  = load_a_frag(hbf + (size_t)m0 * HID + kt * 32, HID, lane);
        bf16x16 b0 = load_b_frag(WhP + ((size_t)nb * 16 + kt) * 512, lane);
        bf16x16 b1 = load_b_frag(WhP + ((size_t)(nb + 64) * 16 + kt) * 512, lane);
        acc0 = wmma_bf16(a, b0, acc0);
        acc1 = wmma_bf16(a, b1, acc1);
      }
      store_c_frag(z + (size_t)m0 * G4 + nb * 16, G4, lane, acc0);
      store_c_frag(z + (size_t)m0 * G4 + (nb + 64) * 16, G4, lane, acc1);
    }
    grid_sync(cnt, gen);

    // ---- F: gates
    for (int idx = gtid; idx < BB * HID; idx += NBLK * NTHR) {
      int b_ = idx >> 9, j = idx & 511;
      const float* zr = z + (size_t)b_ * G4;
      float zi = zr[j], zf = zr[512 + j], zg = zr[1024 + j], zo = zr[1536 + j];
      float cn = sigmoidf(zf) * c[idx] + sigmoidf(zi) * tanhf(zg);
      float hn = sigmoidf(zo) * tanhf(cn);
      c[idx] = cn;
      hbf[idx] = (__bf16)hn;
      Hcat[(size_t)(t * BB + b_) * (2 * HID) + j] = (__bf16)hn;
    }
    grid_sync(cnt, gen);
  }
}

// ---------------------------------------------------------------------------
// In-place log-softmax over 32000-wide rows of d_out.
// ---------------------------------------------------------------------------
__global__ void __launch_bounds__(256) log_softmax_kernel(float* __restrict__ out) {
  __shared__ float red[256];
  float* p = out + (size_t)blockIdx.x * VOUT;
  int tid = threadIdx.x;

  float m = -3.4e38f;
  for (int i = tid; i < VOUT; i += 256) m = fmaxf(m, p[i]);
  red[tid] = m; __syncthreads();
  for (int s = 128; s > 0; s >>= 1) {
    if (tid < s) red[tid] = fmaxf(red[tid], red[tid + s]);
    __syncthreads();
  }
  m = red[0]; __syncthreads();

  float sum = 0.f;
  for (int i = tid; i < VOUT; i += 256) sum += expf(p[i] - m);
  red[tid] = sum; __syncthreads();
  for (int s = 128; s > 0; s >>= 1) {
    if (tid < s) red[tid] += red[tid + s];
    __syncthreads();
  }
  float lse = logf(red[0]) + m;
  for (int i = tid; i < VOUT; i += 256) p[i] = p[i] - lse;
}

// ---------------------------------------------------------------------------
// Host launcher
// ---------------------------------------------------------------------------
extern "C" void kernel_launch(void* const* d_in, const int* in_sizes, int n_in,
                              void* d_out, int out_size, void* d_ws, size_t ws_size,
                              hipStream_t stream) {
  (void)in_sizes; (void)n_in; (void)out_size; (void)ws_size;

  const int*   src       = (const int*)d_in[0];
  const int*   tgt       = (const int*)d_in[1];
  const float* src_embed = (const float*)d_in[2];
  const float* enc_Wx    = (const float*)d_in[3];
  const float* enc_Wh    = (const float*)d_in[4];
  const float* enc_b     = (const float*)d_in[5];
  const float* tgt_embed = (const float*)d_in[6];
  const float* att_Wh    = (const float*)d_in[7];
  const float* att_Ws    = (const float*)d_in[8];
  const float* att_v     = (const float*)d_in[9];
  const float* dec_Wx    = (const float*)d_in[10];
  const float* dec_Wh    = (const float*)d_in[11];
  const float* dec_b     = (const float*)d_in[12];
  const float* h2o_W     = (const float*)d_in[13];
  const float* h2o_b     = (const float*)d_in[14];
  float*       out       = (float*)d_out;

  char* ws = (char*)d_ws;
  size_t off = 0;
  auto alloc = [&](size_t bytes) -> void* {
    void* p = ws + off;
    off = (off + bytes + 255) & ~(size_t)255;
    return p;
  };

  __bf16* pEncWx    = (__bf16*)alloc((size_t)EMB * G4 * 2);
  __bf16* pEncWh    = (__bf16*)alloc((size_t)HID * G4 * 2);
  __bf16* pAttWh    = (__bf16*)alloc((size_t)HID * HID * 2);
  __bf16* pAttWs    = (__bf16*)alloc((size_t)HID * HID * 2);
  __bf16* pDecWxTop = (__bf16*)alloc((size_t)EMB * G4 * 2);
  __bf16* pDecWxBot = (__bf16*)alloc((size_t)HID * G4 * 2);
  __bf16* pDecWh    = (__bf16*)alloc((size_t)HID * G4 * 2);
  __bf16* pH2o      = (__bf16*)alloc((size_t)(2 * HID) * VOUT * 2);
  __bf16* srcEmbB   = (__bf16*)alloc((size_t)NROW * EMB * 2);
  __bf16* tgtEmbB   = (__bf16*)alloc((size_t)NROW * EMB * 2);
  float*  Xenc      = (float*)alloc((size_t)NROW * G4 * 4);
  float*  Xdec      = (float*)alloc((size_t)NROW * G4 * 4);
  float*  encHidF   = (float*)alloc((size_t)NROW * HID * 4);
  __bf16* encHidB   = (__bf16*)alloc((size_t)NROW * HID * 2);
  float*  encT      = (float*)alloc((size_t)NROW * HID * 4);
  float*  cState    = (float*)alloc((size_t)BB * HID * 4);
  __bf16* hbf       = (__bf16*)alloc((size_t)BB * HID * 2);
  float*  zbuf      = (float*)alloc((size_t)BB * G4 * 4);
  float*  decT      = (float*)alloc((size_t)BB * HID * 4);
  float*  attnB     = (float*)alloc((size_t)BB * LSEQ * 4);
  __bf16* ctxbf     = (__bf16*)alloc((size_t)BB * HID * 2);
  __bf16* Hcat      = (__bf16*)alloc((size_t)NROW * 2 * HID * 2);
  int*    syncBuf   = (int*)alloc(256);
  int*    cnt       = syncBuf;
  int*    gen       = syncBuf + 32;

  // --- weight packing + embedding gathers ---
  {
    int tot;
    tot = EMB * G4;
    pack_b_kernel<<<(tot + 255) / 256, 256, 0, stream>>>(enc_Wx, pEncWx, EMB / 32, G4, 0, tot);
    pack_b_kernel<<<(tot + 255) / 256, 256, 0, stream>>>(enc_Wh, pEncWh, HID / 32, G4, 0, tot);
    pack_b_kernel<<<(tot + 255) / 256, 256, 0, stream>>>(dec_Wx, pDecWxTop, EMB / 32, G4, 0, tot);
    pack_b_kernel<<<(tot + 255) / 256, 256, 0, stream>>>(dec_Wx, pDecWxBot, HID / 32, G4, EMB, tot);
    pack_b_kernel<<<(tot + 255) / 256, 256, 0, stream>>>(dec_Wh, pDecWh, HID / 32, G4, 0, tot);
    tot = HID * HID;
    pack_b_kernel<<<(tot + 255) / 256, 256, 0, stream>>>(att_Wh, pAttWh, HID / 32, HID, 0, tot);
    pack_b_kernel<<<(tot + 255) / 256, 256, 0, stream>>>(att_Ws, pAttWs, HID / 32, HID, 0, tot);
    tot = 2 * HID * VOUT;
    pack_b_kernel<<<(tot + 255) / 256, 256, 0, stream>>>(h2o_W, pH2o, (2 * HID) / 32, VOUT, 0, tot);
  }
  gather_embed_kernel<<<NROW, 128, 0, stream>>>(src, src_embed, srcEmbB);
  gather_embed_kernel<<<NROW, 128, 0, stream>>>(tgt, tgt_embed, tgtEmbB);
  (void)hipMemsetAsync(syncBuf, 0, 256, stream);

  // --- hoisted big GEMMs ---
  gemm_bf16_kernel<false><<<dim3(NROW / 128, G4 / 64), 256, 0, stream>>>(
      srcEmbB, pEncWx, enc_b, Xenc, EMB, G4);

  // --- encoder recurrence ---
  encoder_kernel<<<NBLK, NTHR, 0, stream>>>(Xenc, pEncWh, cState, hbf, zbuf,
                                            encHidF, encHidB, cnt, gen);

  // enc_t = enc_hiddens @ att_Wh (decoder-state independent)
  gemm_bf16_kernel<false><<<dim3(NROW / 128, HID / 64), 256, 0, stream>>>(
      encHidB, pAttWh, nullptr, encT, HID, HID);

  // X_dec = tgt_emb @ dec_Wx[:E] + dec_b
  gemm_bf16_kernel<false><<<dim3(NROW / 128, G4 / 64), 256, 0, stream>>>(
      tgtEmbB, pDecWxTop, dec_b, Xdec, EMB, G4);

  (void)hipMemsetAsync(syncBuf, 0, 256, stream);

  // --- decoder recurrence (h,c carry over from encoder buffers) ---
  decoder_kernel<<<NBLK, NTHR, 0, stream>>>(Xdec, pDecWxBot, pDecWh, pAttWs,
                                            encT, encHidF, att_v, cState, hbf,
                                            zbuf, decT, attnB, ctxbf, Hcat,
                                            cnt, gen);

  // --- logits: Hcat (4096x1024) @ h2o_W (1024x32000) -> d_out (row remapped) ---
  gemm_bf16_kernel<true><<<dim3(NROW / 128, VOUT / 64), 256, 0, stream>>>(
      Hcat, pH2o, h2o_b, out, 2 * HID, VOUT);

  // --- in-place log_softmax over vocab ---
  log_softmax_kernel<<<NROW, 256, 0, stream>>>(out);
}
